// Transformer_15934328668485
// MI455X (gfx1250) — matrix operand order repaired and checked by reading, not compiled, not run on hardware
//
#include <hip/hip_runtime.h>
#include <hip/hip_bf16.h>
#include <math.h>
#include <stdint.h>

// ---------------------------------------------------------------------------
// Problem constants (B,S,V,D,F) = (8, 2048, 512, 512, 2048)
// ---------------------------------------------------------------------------
#define BB 8
#define SS 2048
#define DD 512
#define FF 2048
#define MM (BB * SS)            // 16384 rows in all activation GEMMs
#define TOTAL (BB * SS * DD)    // 8388608 elements of [B,S,D]

typedef __attribute__((ext_vector_type(16))) __bf16 v16bf;
typedef __attribute__((ext_vector_type(8)))  __bf16 v8bf;
typedef __attribute__((ext_vector_type(8)))  float  v8f;

static __device__ __forceinline__ v8f wmma_bf16(v16bf a, v16bf b, v8f c) {
  // D = A(16x32 bf16) x B(32x16 bf16) + C(16x16 f32)
  return __builtin_amdgcn_wmma_f32_16x16x32_bf16(false, a, false, b, (short)0, c,
                                                 false, false);
}

// Build a v16bf fragment from two contiguous 8-element (16B) chunks.
static __device__ __forceinline__ v16bf ldfrag(const __bf16* lo, const __bf16* hi) {
  union { v16bf v; v8bf h[2]; } u;
  u.h[0] = *(const v8bf*)lo;
  u.h[1] = *(const v8bf*)hi;
  return u.v;
}

static __device__ __forceinline__ v8f zero8() {
  v8f z = {0.f, 0.f, 0.f, 0.f, 0.f, 0.f, 0.f, 0.f};
  return z;
}

// CDNA5 async global->LDS copy (16B per lane), tracked by ASYNCcnt.
// Generic-pointer low 32 bits are the LDS byte offset (ISA aperture rule).
static __device__ __forceinline__ void async_copy_b128(const __bf16* g, __bf16* l) {
  unsigned lds_off = (unsigned)(uintptr_t)l;
  asm volatile("global_load_async_to_lds_b128 %0, %1, off"
               :: "v"(lds_off), "v"(g) : "memory");
}
static __device__ __forceinline__ void wait_async0() {
  asm volatile("s_wait_asynccnt 0x0" ::: "memory");
}

// ---------------------------------------------------------------------------
// Embedding lookup + sinusoidal positional encoding -> x1 (f32 + bf16)
// ---------------------------------------------------------------------------
__global__ void embed_pe_kernel(const int* __restrict__ x,
                                const float* __restrict__ emb,
                                float* __restrict__ x1f,
                                __bf16* __restrict__ x1b) {
  size_t i = (size_t)blockIdx.x * blockDim.x + threadIdx.x;   // < TOTAL
  int d = (int)(i & (DD - 1));
  size_t bs = i >> 9;                 // b*S + s
  int s = (int)(bs & (SS - 1));
  int tok = x[bs];
  float e = emb[(size_t)tok * DD + d];
  // angle = pos / 10000^(2*d/D)
  float expo = (2.0f * (float)d) * (1.0f / (float)DD);
  float angle = (float)s * expf(-expo * 9.210340371976184f); // ln(10000)
  float pe = (d & 1) ? cosf(angle) : sinf(angle);
  float v = e + pe;
  x1f[i] = v;
  x1b[i] = (__bf16)v;
}

// ---------------------------------------------------------------------------
// Weight transpose + f32 -> bf16:  W[K][N] -> WT[N][K]
// ---------------------------------------------------------------------------
__global__ void wtrans_kernel(const float* __restrict__ w,
                              __bf16* __restrict__ wt, int K, int N) {
  size_t i = (size_t)blockIdx.x * blockDim.x + threadIdx.x;
  if (i >= (size_t)K * N) return;
  int n = (int)(i % N);
  int kk = (int)(i / N);
  wt[(size_t)n * K + kk] = (__bf16)w[i];
}

// ---------------------------------------------------------------------------
// Generic bf16 WMMA GEMM:  C[M,N] = A[M,K] @ B[K,N]  (BT is [N][K] bf16)
// 128x128 workgroup tile, 8 waves, each wave owns a 4x2 grid of 16x16
// subtiles (8 WMMAs / K-step). Double-buffered LDS staged with CDNA5
// async global->LDS copies (ASYNCcnt), one barrier per K-step.
// Epilogue fuses +bias[n], +res (f32), optional ReLU; writes f32 and/or bf16.
// ---------------------------------------------------------------------------
__launch_bounds__(256)
__global__ void gemm_bf16_kernel(const __bf16* __restrict__ A,
                                 const __bf16* __restrict__ BT,
                                 const float* __restrict__ bias,
                                 const float* __restrict__ res,
                                 float* __restrict__ outF,
                                 __bf16* __restrict__ outB,
                                 int M, int N, int K, int relu) {
  __shared__ __align__(16) __bf16 As[2][128][32];   // 16 KB
  __shared__ __align__(16) __bf16 Bs[2][128][32];   // 16 KB
  const int tid = threadIdx.x;
  const int wave = tid >> 5, lane = tid & 31;
  const int h = lane >> 4, l15 = lane & 15;
  const int m0 = blockIdx.y * 128, n0 = blockIdx.x * 128;
  const int wr = wave >> 2, wc = wave & 3;          // 2x4 wave grid

  v8f acc[4][2];
#pragma unroll
  for (int i = 0; i < 4; i++)
#pragma unroll
    for (int j = 0; j < 2; j++) acc[i][j] = zero8();

  // Tile staging: 128x32 bf16 = 512 16B-chunks per matrix; 2 per thread.
  const int r0 = tid >> 2;               // rows 0..63 (and +64)
  const int c0 = (tid & 3) * 8;          // col 0,8,16,24

  const int nsteps = K >> 5;             // K/32

  // prologue: stage 0 into buffer 0
  {
    const int kb = 0;
    async_copy_b128(&A [(size_t)(m0 + r0)      * K + kb + c0], &As[0][r0][c0]);
    async_copy_b128(&A [(size_t)(m0 + r0 + 64) * K + kb + c0], &As[0][r0 + 64][c0]);
    async_copy_b128(&BT[(size_t)(n0 + r0)      * K + kb + c0], &Bs[0][r0][c0]);
    async_copy_b128(&BT[(size_t)(n0 + r0 + 64) * K + kb + c0], &Bs[0][r0 + 64][c0]);
  }
  wait_async0();
  __syncthreads();

  for (int s = 0; s < nsteps; s++) {
    const int sel = s & 1;
    if (s + 1 < nsteps) {               // overlap next-stage copies with math
      const int kb = (s + 1) << 5;
      const int nx = sel ^ 1;
      async_copy_b128(&A [(size_t)(m0 + r0)      * K + kb + c0], &As[nx][r0][c0]);
      async_copy_b128(&A [(size_t)(m0 + r0 + 64) * K + kb + c0], &As[nx][r0 + 64][c0]);
      async_copy_b128(&BT[(size_t)(n0 + r0)      * K + kb + c0], &Bs[nx][r0][c0]);
      async_copy_b128(&BT[(size_t)(n0 + r0 + 64) * K + kb + c0], &Bs[nx][r0 + 64][c0]);
    }

    v16bf afr[4], bfr[2];
#pragma unroll
    for (int i = 0; i < 4; i++) {
      const int row = (wr * 4 + i) * 16 + l15;
      afr[i] = ldfrag(&As[sel][row][8 * h], &As[sel][row][16 + 8 * h]);
    }
#pragma unroll
    for (int j = 0; j < 2; j++) {
      const int col = (wc * 2 + j) * 16 + l15;
      bfr[j] = ldfrag(&Bs[sel][col][8 * h], &Bs[sel][col][16 + 8 * h]);
    }
#pragma unroll
    for (int i = 0; i < 4; i++)
#pragma unroll
      for (int j = 0; j < 2; j++)
        acc[i][j] = wmma_bf16(afr[i], bfr[j], acc[i][j]);

    wait_async0();                      // next buffer fully in LDS (this wave)
    __syncthreads();                    // ... and for all waves; readers done
  }

#pragma unroll
  for (int i = 0; i < 4; i++) {
#pragma unroll
    for (int j = 0; j < 2; j++) {
      const int n = n0 + (wc * 2 + j) * 16 + l15;
      const float bn = bias ? bias[n] : 0.0f;
#pragma unroll
      for (int r = 0; r < 8; r++) {
        const int m = m0 + (wr * 4 + i) * 16 + 8 * h + r;
        float c = acc[i][j][r] + bn;
        const size_t idx = (size_t)m * N + n;
        if (res) c += res[idx];
        if (relu) c = fmaxf(c, 0.0f);
        if (outF) outF[idx] = c;
        if (outB) outB[idx] = (__bf16)c;
      }
    }
  }
}

// ---------------------------------------------------------------------------
// Flash-style full-D attention.
//   grid = (S/16 query tiles, B); block = 512 threads = 16 waves.
//   scores contract over D=512; softmax online; P @ V contract over S.
// ---------------------------------------------------------------------------
__launch_bounds__(512)
__global__ void attn_kernel(const __bf16* __restrict__ Qm,
                            const __bf16* __restrict__ Km,
                            const __bf16* __restrict__ Vm,
                            __bf16* __restrict__ Om) {
  __shared__ __align__(16) __bf16 Qs[16][512];   // 16 KB, pre-scaled queries
  __shared__ __align__(16) __bf16 Ps[16][256];   //  8 KB, probabilities (A-layout)
  __shared__ __align__(16) __bf16 Vt[512][32];   // 32 KB, transposed V chunk
  __shared__ float rmax[16][16];                 // [wave][row]
  __shared__ float rsum[16][16];
  __shared__ float mrun[16], lrun[16], scf[16];

  const int tid = threadIdx.x;
  const int wave = tid >> 5, lane = tid & 31;
  const int h = lane >> 4, l15 = lane & 15;
  const int b = blockIdx.y;
  const int q0 = blockIdx.x * 16;
  const float kscale = 0.044194173824159216f;    // 1/sqrt(512)

  // Stage Q tile (scaled) into LDS: 1024 chunks of 8 -> 2 per thread.
  for (int i = tid; i < 1024; i += 512) {
    int r = i >> 6, c8 = (i & 63) * 8;
    v8bf q8 = *(const v8bf*)&Qm[((size_t)b * SS + q0 + r) * DD + c8];
    v8bf o8;
#pragma unroll
    for (int j = 0; j < 8; j++) o8[j] = (__bf16)((float)q8[j] * kscale);
    *(v8bf*)&Qs[r][c8] = o8;
  }
  if (tid < 16) { mrun[tid] = -3.0e38f; lrun[tid] = 0.0f; }
  __syncthreads();

  v8f acc0 = zero8(), acc1 = zero8();            // two 16-wide output col tiles
  const int nA = wave * 32, nB = nA + 16;

  for (int kb = 0; kb < SS; kb += 256) {
    // ---- scores: this wave handles keys [kb+16*wave, kb+16*wave+16) ----
    v8f sacc = zero8();
    const int key = kb + wave * 16 + l15;
    const __bf16* krow = Km + ((size_t)b * SS + key) * DD;
#pragma unroll 4
    for (int d0 = 0; d0 < DD; d0 += 32) {
      v16bf af = ldfrag(&Qs[l15][d0 + 8 * h], &Qs[l15][d0 + 16 + 8 * h]);
      v16bf bf = ldfrag(&krow[d0 + 8 * h], &krow[d0 + 16 + 8 * h]);
      sacc = wmma_bf16(af, bf, sacc);
    }
    __syncthreads();            // prior Ps/Vt consumers done; stats reusable

    // ---- per-row max of this wave's 16-key subtile ----
#pragma unroll
    for (int r = 0; r < 8; r++) {
      float x = sacc[r];
      for (int off = 8; off; off >>= 1) x = fmaxf(x, __shfl_xor(x, off, 16));
      if (l15 == 0) rmax[wave][8 * h + r] = x;
    }
    __syncthreads();

    if (tid < 16) {             // combine 16 waves -> running max + scale
      float mb = rmax[0][tid];
#pragma unroll
      for (int w = 1; w < 16; w++) mb = fmaxf(mb, rmax[w][tid]);
      float mo = mrun[tid];
      float mn = fmaxf(mo, mb);
      scf[tid] = __expf(mo - mn);
      mrun[tid] = mn;
    }
    __syncthreads();

    // ---- P = exp(S - m); write bf16 P; row sums; rescale accumulators ----
#pragma unroll
    for (int r = 0; r < 8; r++) {
      const int row = 8 * h + r;
      float p = __expf(sacc[r] - mrun[row]);
      Ps[row][wave * 16 + l15] = (__bf16)p;
      float sm = p;
      for (int off = 8; off; off >>= 1) sm += __shfl_xor(sm, off, 16);
      if (l15 == 0) rsum[wave][row] = sm;
      const float s = scf[row];
      acc0[r] *= s;
      acc1[r] *= s;
    }
    __syncthreads();

    if (tid < 16) {             // running denominator
      float sb = 0.0f;
#pragma unroll
      for (int w = 0; w < 16; w++) sb += rsum[w][tid];
      lrun[tid] = lrun[tid] * scf[tid] + sb;
    }

    // ---- P @ V over this 256-key block, 32 keys per staged chunk ----
    for (int kc = 0; kc < 256; kc += 32) {
      __syncthreads();          // Ps ready / previous Vt consumers done
      for (int i = tid; i < 32 * 64; i += 512) {  // 2048 chunks of 8
        int kk = i >> 6;
        int c8 = (i & 63) * 8;
        v8bf r8 = *(const v8bf*)&Vm[((size_t)b * SS + kb + kc + kk) * DD + c8];
#pragma unroll
        for (int j = 0; j < 8; j++) Vt[c8 + j][kk] = r8[j];
      }
      __syncthreads();
      v16bf pf = ldfrag(&Ps[l15][kc + 8 * h], &Ps[l15][kc + 16 + 8 * h]);
      v16bf v0 = ldfrag(&Vt[nA + l15][8 * h], &Vt[nA + l15][16 + 8 * h]);
      acc0 = wmma_bf16(pf, v0, acc0);
      v16bf v1 = ldfrag(&Vt[nB + l15][8 * h], &Vt[nB + l15][16 + 8 * h]);
      acc1 = wmma_bf16(pf, v1, acc1);
    }
  }

  // ---- normalize by softmax denominator, emit bf16 attention output ----
#pragma unroll
  for (int r = 0; r < 8; r++) {
    const int row = 8 * h + r;
    const float linv = 1.0f / lrun[row];
    const size_t base = ((size_t)b * SS + q0 + row) * DD;
    Om[base + nA + l15] = (__bf16)(acc0[r] * linv);
    Om[base + nB + l15] = (__bf16)(acc1[r] * linv);
  }
}

// ---------------------------------------------------------------------------
// Global (whole-tensor) LayerNorm: partial reduce -> finalize -> apply
// ---------------------------------------------------------------------------
__global__ void reduce_partial_kernel(const float* __restrict__ t,
                                      float* __restrict__ part, int n) {
  __shared__ float ss[256], sq[256];
  const int tid = threadIdx.x;
  float s = 0.f, q = 0.f;
  for (size_t i = (size_t)blockIdx.x * 256 + tid; i < (size_t)n;
       i += (size_t)gridDim.x * 256) {
    float v = t[i];
    s += v;
    q += v * v;
  }
  ss[tid] = s; sq[tid] = q;
  __syncthreads();
  for (int off = 128; off; off >>= 1) {
    if (tid < off) { ss[tid] += ss[tid + off]; sq[tid] += sq[tid + off]; }
    __syncthreads();
  }
  if (tid == 0) { part[2 * blockIdx.x] = ss[0]; part[2 * blockIdx.x + 1] = sq[0]; }
}

__global__ void reduce_final_kernel(const float* __restrict__ part,
                                    float* __restrict__ stats, int nblk, float n) {
  __shared__ float ss[256], sq[256];
  const int tid = threadIdx.x;
  float s = 0.f, q = 0.f;
  for (int i = tid; i < nblk; i += 256) { s += part[2 * i]; q += part[2 * i + 1]; }
  ss[tid] = s; sq[tid] = q;
  __syncthreads();
  for (int off = 128; off; off >>= 1) {
    if (tid < off) { ss[tid] += ss[tid + off]; sq[tid] += sq[tid + off]; }
    __syncthreads();
  }
  if (tid == 0) {
    float mean = ss[0] / n;
    float var = (sq[0] - ss[0] * ss[0] / n) / (n - 1.0f);   // ddof = 1
    stats[0] = mean;
    stats[1] = 1.0f / (sqrtf(var) + 1e-9f);
  }
}

__global__ void ln_apply_kernel(const float* __restrict__ t,
                                const float* __restrict__ stats,
                                const float* __restrict__ gamma,
                                const float* __restrict__ beta,
                                float* __restrict__ outF,
                                __bf16* __restrict__ outB) {
  size_t i = (size_t)blockIdx.x * blockDim.x + threadIdx.x;   // < TOTAL
  const int d = (int)(i & (DD - 1));
  const float mean = stats[0], inv = stats[1];
  float v = gamma[d] * ((t[i] - mean) * inv) + beta[d];
  outF[i] = v;
  if (outB) outB[i] = (__bf16)v;
}

// ---------------------------------------------------------------------------
// Orchestration
// ---------------------------------------------------------------------------
extern "C" void kernel_launch(void* const* d_in, const int* in_sizes, int n_in,
                              void* d_out, int out_size, void* d_ws, size_t ws_size,
                              hipStream_t stream) {
  (void)in_sizes; (void)n_in; (void)out_size; (void)ws_size;
  const int*   x     = (const int*)  d_in[0];
  const float* emb   = (const float*)d_in[1];
  const float* Wq    = (const float*)d_in[2];
  const float* bq    = (const float*)d_in[3];
  const float* Wk    = (const float*)d_in[4];
  const float* bk    = (const float*)d_in[5];
  const float* Wv    = (const float*)d_in[6];
  const float* bv    = (const float*)d_in[7];
  const float* Wo    = (const float*)d_in[8];
  const float* bo    = (const float*)d_in[9];
  const float* W1    = (const float*)d_in[10];
  const float* b1    = (const float*)d_in[11];
  const float* W2    = (const float*)d_in[12];
  const float* b2    = (const float*)d_in[13];
  const float* gamma = (const float*)d_in[14];
  const float* beta  = (const float*)d_in[15];
  float* out = (float*)d_out;

  char* w = (char*)d_ws;
  const size_t MB1 = 1ull << 20;
  float*  x1f  = (float*) (w + 0 * MB1);      // 32 MB, x1 f32 (residual)
  __bf16* x1b  = (__bf16*)(w + 32 * MB1);     // 16 MB, x1 bf16
  __bf16* qb   = (__bf16*)(w + 48 * MB1);     // 16 MB
  __bf16* kbuf = (__bf16*)(w + 64 * MB1);     // 16 MB
  __bf16* vbuf = (__bf16*)(w + 80 * MB1);     // 16 MB
  __bf16* ab   = (__bf16*)(w + 96 * MB1);     // 16 MB, attention out bf16
  float*  t1   = (float*) (w + 112 * MB1);    // 32 MB, x1+x2 (reused as x3+x4)
  float*  x3f  = (float*) (w + 144 * MB1);    // 32 MB
  __bf16* x3b  = (__bf16*)(w + 176 * MB1);    // 16 MB
  __bf16* h1   = (__bf16*)(w + 48 * MB1);     // 64 MB, reuses dead q/k/v/attn
  __bf16* WqT  = (__bf16*)(w + 192 * MB1);    // 6 MB of bf16 weights follow
  __bf16* WkT  = WqT + 512 * 512;
  __bf16* WvT  = WkT + 512 * 512;
  __bf16* WoT  = WvT + 512 * 512;
  __bf16* W1T  = WoT + 512 * 512;             // [2048][512]
  __bf16* W2T  = W1T + 512 * 2048;            // [512][2048]
  float*  part = (float*) (w + 200 * MB1);    // 1024 * 2 floats
  float*  stats = part + 2048;                // 2 floats

  // 1. embedding + positional encoding
  embed_pe_kernel<<<TOTAL / 256, 256, 0, stream>>>(x, emb, x1f, x1b);

  // 2. weight transpose/convert to bf16
  wtrans_kernel<<<(512 * 512) / 256, 256, 0, stream>>>(Wq, WqT, 512, 512);
  wtrans_kernel<<<(512 * 512) / 256, 256, 0, stream>>>(Wk, WkT, 512, 512);
  wtrans_kernel<<<(512 * 512) / 256, 256, 0, stream>>>(Wv, WvT, 512, 512);
  wtrans_kernel<<<(512 * 512) / 256, 256, 0, stream>>>(Wo, WoT, 512, 512);
  wtrans_kernel<<<(512 * 2048) / 256, 256, 0, stream>>>(W1, W1T, 512, 2048);
  wtrans_kernel<<<(2048 * 512) / 256, 256, 0, stream>>>(W2, W2T, 2048, 512);

  // 3. Q, K, V projections (bf16 out)
  dim3 g512(512 / 128, MM / 128);             // (4, 128)
  gemm_bf16_kernel<<<g512, 256, 0, stream>>>(x1b, WqT, bq, nullptr, nullptr, qb,
                                             MM, 512, 512, 0);
  gemm_bf16_kernel<<<g512, 256, 0, stream>>>(x1b, WkT, bk, nullptr, nullptr, kbuf,
                                             MM, 512, 512, 0);
  gemm_bf16_kernel<<<g512, 256, 0, stream>>>(x1b, WvT, bv, nullptr, nullptr, vbuf,
                                             MM, 512, 512, 0);

  // 4. attention (softmax(QK^T/sqrt(D)) V)
  attn_kernel<<<dim3(SS / 16, BB), 512, 0, stream>>>(qb, kbuf, vbuf, ab);

  // 5. output projection + residual -> t1 = x1 + x2
  gemm_bf16_kernel<<<g512, 256, 0, stream>>>(ab, WoT, bo, x1f, t1, nullptr,
                                             MM, 512, 512, 0);

  // 6. global layernorm -> x3 (f32 + bf16)
  reduce_partial_kernel<<<1024, 256, 0, stream>>>(t1, part, TOTAL);
  reduce_final_kernel<<<1, 256, 0, stream>>>(part, stats, 1024, (float)TOTAL);
  ln_apply_kernel<<<TOTAL / 256, 256, 0, stream>>>(t1, stats, gamma, beta, x3f, x3b);

  // 7. FFN: h1 = relu(x3 @ W1 + b1); t2 = h1 @ W2 + b2 + x3
  dim3 gF(FF / 128, MM / 128);                // (16, 128)
  gemm_bf16_kernel<<<gF, 256, 0, stream>>>(x3b, W1T, b1, nullptr, nullptr, h1,
                                           MM, FF, 512, 1);
  gemm_bf16_kernel<<<g512, 256, 0, stream>>>(h1, W2T, b2, x3f, t1, nullptr,
                                             MM, 512, FF, 0);

  // 8. final global layernorm -> d_out
  reduce_partial_kernel<<<1024, 256, 0, stream>>>(t1, part, TOTAL);
  reduce_final_kernel<<<1, 256, 0, stream>>>(part, stats, 1024, (float)TOTAL);
  ln_apply_kernel<<<TOTAL / 256, 256, 0, stream>>>(t1, stats, gamma, beta, out,
                                                   nullptr);
}
